// DecoderBlock_50869592655693
// MI455X (gfx1250) — compile-verified, matrix operand order checked by reading, not run on hardware
//
#include <hip/hip_runtime.h>

// ---------------- types ----------------
typedef __bf16 bf16_t;
typedef __bf16 v16bf __attribute__((ext_vector_type(16)));
typedef float  v8f   __attribute__((ext_vector_type(8)));

#define D_MODEL 1024
#define D_FF    4096
#define N_HEADS 16
#define DH      64
#define SEQ     1024
#define BATCH   4
#define ROWS    (BATCH * SEQ)   // 4096

static __device__ inline bf16_t f2bf(float f) {
    unsigned u = __builtin_bit_cast(unsigned, f);
    unsigned r = u + 0x7fffu + ((u >> 16) & 1u);
    unsigned short h = (unsigned short)(r >> 16);
    return __builtin_bit_cast(bf16_t, h);
}

// ---------------------------------------------------------------------------
// Pack fp32 row-major A (MxK) into WMMA A-fragment layout:
//   frag f = mt*Ktiles + kt, lane l holds row m = mt*16 + (l&15),
//   K values: base kt*32 + (l>>4)*8, k = kb + (j<8 ? j : j+8)
// Per-(blockIdx.y) source offset = (bz/nInner)*strideOuter + (bz%nInner)*strideInner.
// ---------------------------------------------------------------------------
__global__ __launch_bounds__(256)
void pack_a_kernel(const float* __restrict__ src, bf16_t* __restrict__ dst,
                   int Mtiles, int Ktiles, int ld, int nInner,
                   long strideOuter, long strideInner, long dstStride) {
    int nf  = Mtiles * Ktiles;
    int tid = blockIdx.x * blockDim.x + threadIdx.x;
    if (tid >= nf * 32) return;
    int f = tid >> 5, l = tid & 31;
    int mt = f / Ktiles, kt = f % Ktiles;
    int bz = blockIdx.y;
    const float* s = src + (size_t)(bz / nInner) * strideOuter
                         + (size_t)(bz % nInner) * strideInner;
    bf16_t* d = dst + (size_t)bz * dstStride + ((size_t)f * 512 + l * 16);
    int m  = mt * 16 + (l & 15);
    int kb = kt * 32 + ((l >> 4) * 8);
    v16bf v;
#pragma unroll
    for (int j = 0; j < 16; j++) {
        int k = kb + (j < 8 ? j : j + 8);
        v[j] = f2bf(s[(size_t)m * ld + k]);
    }
    *(v16bf*)d = v;
}

// Pack fp32 row-major B (KxN) into WMMA B-fragment layout (lane = column).
__global__ __launch_bounds__(256)
void pack_b_kernel(const float* __restrict__ src, bf16_t* __restrict__ dst,
                   int Ntiles, int Ktiles, int ld, int nInner,
                   long strideOuter, long strideInner, long dstStride) {
    int nf  = Ntiles * Ktiles;
    int tid = blockIdx.x * blockDim.x + threadIdx.x;
    if (tid >= nf * 32) return;
    int f = tid >> 5, l = tid & 31;
    int nt = f / Ktiles, kt = f % Ktiles;
    int bz = blockIdx.y;
    const float* s = src + (size_t)(bz / nInner) * strideOuter
                         + (size_t)(bz % nInner) * strideInner;
    bf16_t* d = dst + (size_t)bz * dstStride + ((size_t)f * 512 + l * 16);
    int n  = nt * 16 + (l & 15);
    int kb = kt * 32 + ((l >> 4) * 8);
    v16bf v;
#pragma unroll
    for (int j = 0; j < 16; j++) {
        int k = kb + (j < 8 ? j : j + 8);
        v[j] = f2bf(s[(size_t)k * ld + n]);
    }
    *(v16bf*)d = v;
}

// ---------------------------------------------------------------------------
// GEMM: C(MxN fp32) = A(packed bf16 frags) @ B(packed bf16 frags) [+bias][relu]
// Block = 256 threads = 8 waves arranged 2(m) x 4(n); block tile 128x128.
// Wave tile 64x32 = 4x2 WMMA tiles -> 8 WMMAs per 6 fragment loads per K-step.
// Kt = K/32 fragment count along K.
// ---------------------------------------------------------------------------
__global__ __launch_bounds__(256)
void gemm_bf16_kernel(const bf16_t* __restrict__ Ap, const bf16_t* __restrict__ Bp,
                      float* __restrict__ C, const float* __restrict__ bias,
                      int N, int Kt, int relu) {
    int w = threadIdx.x >> 5, l = threadIdx.x & 31;
    int mt0 = blockIdx.y * 8 + (w & 1) * 4;   // first of 4 m-tiles for this wave
    int nt0 = blockIdx.x * 8 + (w >> 1) * 2;  // first of 2 n-tiles for this wave

    const bf16_t* a0 = Ap + (size_t)(mt0 + 0) * Kt * 512 + l * 16;
    const bf16_t* a1 = Ap + (size_t)(mt0 + 1) * Kt * 512 + l * 16;
    const bf16_t* a2 = Ap + (size_t)(mt0 + 2) * Kt * 512 + l * 16;
    const bf16_t* a3 = Ap + (size_t)(mt0 + 3) * Kt * 512 + l * 16;
    const bf16_t* b0 = Bp + (size_t)(nt0 + 0) * Kt * 512 + l * 16;
    const bf16_t* b1 = Bp + (size_t)(nt0 + 1) * Kt * 512 + l * 16;

    v8f acc[4][2] = {};
    for (int kt = 0; kt < Kt; kt++) {
        v16bf vb0 = *(const v16bf*)b0;  b0 += 512;
        v16bf vb1 = *(const v16bf*)b1;  b1 += 512;
        v16bf va0 = *(const v16bf*)a0;  a0 += 512;
        v16bf va1 = *(const v16bf*)a1;  a1 += 512;
        v16bf va2 = *(const v16bf*)a2;  a2 += 512;
        v16bf va3 = *(const v16bf*)a3;  a3 += 512;
        acc[0][0] = __builtin_amdgcn_wmma_f32_16x16x32_bf16(false, va0, false, vb0, (short)0, acc[0][0], false, false);
        acc[0][1] = __builtin_amdgcn_wmma_f32_16x16x32_bf16(false, va0, false, vb1, (short)0, acc[0][1], false, false);
        acc[1][0] = __builtin_amdgcn_wmma_f32_16x16x32_bf16(false, va1, false, vb0, (short)0, acc[1][0], false, false);
        acc[1][1] = __builtin_amdgcn_wmma_f32_16x16x32_bf16(false, va1, false, vb1, (short)0, acc[1][1], false, false);
        acc[2][0] = __builtin_amdgcn_wmma_f32_16x16x32_bf16(false, va2, false, vb0, (short)0, acc[2][0], false, false);
        acc[2][1] = __builtin_amdgcn_wmma_f32_16x16x32_bf16(false, va2, false, vb1, (short)0, acc[2][1], false, false);
        acc[3][0] = __builtin_amdgcn_wmma_f32_16x16x32_bf16(false, va3, false, vb0, (short)0, acc[3][0], false, false);
        acc[3][1] = __builtin_amdgcn_wmma_f32_16x16x32_bf16(false, va3, false, vb1, (short)0, acc[3][1], false, false);
    }

    int col = l & 15, rb = (l >> 4) * 8;
#pragma unroll
    for (int i = 0; i < 4; i++) {
#pragma unroll
        for (int j = 0; j < 2; j++) {
            int n = (nt0 + j) * 16 + col;
            float bv = bias ? bias[n] : 0.0f;
#pragma unroll
            for (int r = 0; r < 8; r++) {
                int m = (mt0 + i) * 16 + rb + r;
                float v = acc[i][j][r] + bv;
                if (relu) v = fmaxf(v, 0.0f);
                C[(size_t)m * N + n] = v;
            }
        }
    }
}

// ---------------------------------------------------------------------------
// Fused attention: one block = 16 query rows of one (b,h).
// Phase 1: scores (16x1024) = Q K^T / sqrt(Dh) (+causal mask) -> LDS (fp32)
// Phase 2: row softmax (wave32 shuffle reductions)
// Phase 3: out (16x64) = P @ V  (P fragments built from LDS, fp32->bf16)
// ---------------------------------------------------------------------------
__global__ __launch_bounds__(256)
void attn_kernel(const bf16_t* __restrict__ Qp, const bf16_t* __restrict__ Kp,
                 const bf16_t* __restrict__ Vp, float* __restrict__ out, int causal) {
    extern __shared__ float sc[];   // [16][SEQ] = 64 KB
    int qt = blockIdx.x;            // query tile (0..63)
    int bh = blockIdx.y;            // b*N_HEADS + h
    int b = bh >> 4, h = bh & 15;
    int w = threadIdx.x >> 5, l = threadIdx.x & 31;
    int col = l & 15, rb = (l >> 4) * 8;

    // ---- Phase 1: scores ----
    const bf16_t* qbase = Qp + ((size_t)bh * 64 + qt) * 2 * 512 + l * 16;
    v16bf qa0 = *(const v16bf*)(qbase);
    v16bf qa1 = *(const v16bf*)(qbase + 512);
    for (int i = 0; i < 8; i++) {
        int nt = w * 8 + i;     // key tile
        const bf16_t* kb_ = Kp + ((size_t)bh * 64 + nt) * 2 * 512 + l * 16;
        v16bf k0 = *(const v16bf*)(kb_);
        v16bf k1 = *(const v16bf*)(kb_ + 512);
        v8f acc = {};
        acc = __builtin_amdgcn_wmma_f32_16x16x32_bf16(false, qa0, false, k0, (short)0, acc, false, false);
        acc = __builtin_amdgcn_wmma_f32_16x16x32_bf16(false, qa1, false, k1, (short)0, acc, false, false);
        int nn = nt * 16 + col;
#pragma unroll
        for (int r = 0; r < 8; r++) {
            int m  = rb + r;
            int ss = qt * 16 + m;
            float v = acc[r] * 0.125f;          // 1/sqrt(64)
            if (causal && nn > ss) v = -1e9f;
            sc[m * SEQ + nn] = v;
        }
    }
    __syncthreads();

    // ---- Phase 2: softmax (wave w handles rows 2w, 2w+1) ----
    for (int rr = 0; rr < 2; rr++) {
        float* p = sc + (w * 2 + rr) * SEQ;
        float mx = -3.4e38f;
        for (int i = l; i < SEQ; i += 32) mx = fmaxf(mx, p[i]);
        for (int o = 16; o > 0; o >>= 1) mx = fmaxf(mx, __shfl_xor(mx, o, 32));
        float s = 0.0f;
        for (int i = l; i < SEQ; i += 32) { float e = __expf(p[i] - mx); p[i] = e; s += e; }
        for (int o = 16; o > 0; o >>= 1) s += __shfl_xor(s, o, 32);
        float inv = 1.0f / s;
        for (int i = l; i < SEQ; i += 32) p[i] *= inv;
    }
    __syncthreads();

    // ---- Phase 3: P @ V (waves 0..3, one dh-tile each) ----
    if (w < 4) {
        int dt = w;
        v8f acc = {};
        int m   = l & 15;
        int kb0 = (l >> 4) * 8;
        for (int kt = 0; kt < 32; kt++) {
            v16bf av;
            int kb = kt * 32 + kb0;
#pragma unroll
            for (int j = 0; j < 16; j++) {
                int k = kb + (j < 8 ? j : j + 8);
                av[j] = f2bf(sc[m * SEQ + k]);
            }
            v16bf bv = *(const v16bf*)(Vp + (((size_t)bh * 4 + dt) * 32 + kt) * 512 + l * 16);
            acc = __builtin_amdgcn_wmma_f32_16x16x32_bf16(false, av, false, bv, (short)0, acc, false, false);
        }
#pragma unroll
        for (int r = 0; r < 8; r++) {
            int srow = qt * 16 + rb + r;
            out[((size_t)b * SEQ + srow) * D_MODEL + h * DH + dt * 16 + col] = acc[r];
        }
    }
}

// ---------------------------------------------------------------------------
// out[row] = resid[row] + LayerNorm(src[row]) * g + b   (D_MODEL = 1024)
// ---------------------------------------------------------------------------
__global__ __launch_bounds__(256)
void ln_residual_kernel(const float* __restrict__ src, const float* __restrict__ resid,
                        const float* __restrict__ g, const float* __restrict__ bb,
                        float* __restrict__ out) {
    int row = blockIdx.x;
    const float* x  = src   + (size_t)row * D_MODEL;
    const float* rz = resid + (size_t)row * D_MODEL;
    float*       o  = out   + (size_t)row * D_MODEL;
    int t = threadIdx.x;
    float v[4]; float s = 0.0f, s2 = 0.0f;
#pragma unroll
    for (int i = 0; i < 4; i++) {
        int e = t + i * 256;
        v[i] = x[e]; s += v[i]; s2 += v[i] * v[i];
    }
    for (int ofs = 16; ofs > 0; ofs >>= 1) { s += __shfl_xor(s, ofs, 32); s2 += __shfl_xor(s2, ofs, 32); }
    __shared__ float rs[8], rs2[8];
    int w = t >> 5, l = t & 31;
    if (l == 0) { rs[w] = s; rs2[w] = s2; }
    __syncthreads();
    if (t == 0) {
        float a = 0.0f, c = 0.0f;
        for (int j = 0; j < 8; j++) { a += rs[j]; c += rs2[j]; }
        rs[0] = a; rs2[0] = c;
    }
    __syncthreads();
    float mu   = rs[0] * (1.0f / D_MODEL);
    float var  = rs2[0] * (1.0f / D_MODEL) - mu * mu;
    float rstd = rsqrtf(var + 1e-5f);
#pragma unroll
    for (int i = 0; i < 4; i++) {
        int e = t + i * 256;
        o[e] = rz[e] + (v[i] - mu) * rstd * g[e] + bb[e];
    }
}

// ---------------------------------------------------------------------------
// Host orchestration
// ---------------------------------------------------------------------------
static void launch_pack_a(const float* s, bf16_t* d, int Mt, int Kt, int ld,
                          int nInner, long so, long si, long ds, int nz, hipStream_t st) {
    int threads = Mt * Kt * 32;
    dim3 grid((threads + 255) / 256, nz);
    pack_a_kernel<<<grid, 256, 0, st>>>(s, d, Mt, Kt, ld, nInner, so, si, ds);
}
static void launch_pack_b(const float* s, bf16_t* d, int Nt, int Kt, int ld,
                          int nInner, long so, long si, long ds, int nz, hipStream_t st) {
    int threads = Nt * Kt * 32;
    dim3 grid((threads + 255) / 256, nz);
    pack_b_kernel<<<grid, 256, 0, st>>>(s, d, Nt, Kt, ld, nInner, so, si, ds);
}
static void launch_gemm(const bf16_t* A, const bf16_t* B, float* C, const float* bias,
                        int M, int N, int K, int relu, hipStream_t st) {
    dim3 grid(N / 128, M / 128);
    gemm_bf16_kernel<<<grid, 256, 0, st>>>(A, B, C, bias, N, K / 32, relu);
}

extern "C" void kernel_launch(void* const* d_in, const int* in_sizes, int n_in,
                              void* d_out, int out_size, void* d_ws, size_t ws_size,
                              hipStream_t stream) {
    const float* X    = (const float*)d_in[0];
    const float* Wq1  = (const float*)d_in[1];
    const float* Wk1  = (const float*)d_in[2];
    const float* Wv1  = (const float*)d_in[3];
    const float* Wo1  = (const float*)d_in[4];
    const float* Wq2  = (const float*)d_in[5];
    const float* Wk2  = (const float*)d_in[6];
    const float* Wv2  = (const float*)d_in[7];
    const float* Wo2  = (const float*)d_in[8];
    const float* ln_g = (const float*)d_in[9];
    const float* ln_b = (const float*)d_in[10];
    const float* W1   = (const float*)d_in[11];
    const float* b1   = (const float*)d_in[12];
    const float* W2   = (const float*)d_in[13];
    const float* b2   = (const float*)d_in[14];

    char* ws = (char*)d_ws;
    // workspace layout (bytes): [0,32M) packed weights | [32M,64M) packed-A/QKV frags
    // [64M,128M) fp32 scratch | [128M,144M) act1 | [144M,160M) act2
    bf16_t* wp   = (bf16_t*)(ws);
    bf16_t* ap   = (bf16_t*)(ws + (32u << 20));
    float*  t0   = (float*)(ws + (64u << 20));
    float*  act1 = (float*)(ws + (128u << 20));
    float*  act2 = (float*)(ws + (144u << 20));

    const size_t WSZ = 1u << 20;                 // 1024*1024 elements per attn weight
    bf16_t* Wq1p = wp + 0 * WSZ;  bf16_t* Wk1p = wp + 1 * WSZ;
    bf16_t* Wv1p = wp + 2 * WSZ;  bf16_t* Wo1p = wp + 3 * WSZ;
    bf16_t* Wq2p = wp + 4 * WSZ;  bf16_t* Wk2p = wp + 5 * WSZ;
    bf16_t* Wv2p = wp + 6 * WSZ;  bf16_t* Wo2p = wp + 7 * WSZ;
    bf16_t* W1p  = wp + 8 * WSZ;                 // 1024x4096
    bf16_t* W2p  = wp + 12 * WSZ;                // 4096x1024

    // ---- pack weights (B-operand layout) ----
    launch_pack_b(Wq1, Wq1p, 64, 32, 1024, 1, 0, 0, 0, 1, stream);
    launch_pack_b(Wk1, Wk1p, 64, 32, 1024, 1, 0, 0, 0, 1, stream);
    launch_pack_b(Wv1, Wv1p, 64, 32, 1024, 1, 0, 0, 0, 1, stream);
    launch_pack_b(Wo1, Wo1p, 64, 32, 1024, 1, 0, 0, 0, 1, stream);
    launch_pack_b(Wq2, Wq2p, 64, 32, 1024, 1, 0, 0, 0, 1, stream);
    launch_pack_b(Wk2, Wk2p, 64, 32, 1024, 1, 0, 0, 0, 1, stream);
    launch_pack_b(Wv2, Wv2p, 64, 32, 1024, 1, 0, 0, 0, 1, stream);
    launch_pack_b(Wo2, Wo2p, 64, 32, 1024, 1, 0, 0, 0, 1, stream);
    launch_pack_b(W1, W1p, 256, 32, 4096, 1, 0, 0, 0, 1, stream);
    launch_pack_b(W2, W2p, 64, 128, 1024, 1, 0, 0, 0, 1, stream);

    float* Qf = t0;
    float* Kf = t0 + (size_t)4 * 1024 * 1024;
    float* Vf = t0 + (size_t)8 * 1024 * 1024;
    bf16_t* Qp = ap;
    bf16_t* Kp = ap + (size_t)4 * 1024 * 1024;
    bf16_t* Vp = ap + (size_t)8 * 1024 * 1024;

    auto run_attn_layer = [&](const float* in, const bf16_t* wqp, const bf16_t* wkp,
                              const bf16_t* wvp, const bf16_t* wop, float* outProj, int causal) {
        // project Q,K,V
        launch_pack_a(in, ap, ROWS / 16, 32, 1024, 1, 0, 0, 0, 1, stream);
        launch_gemm(ap, wqp, Qf, nullptr, ROWS, 1024, 1024, 0, stream);
        launch_gemm(ap, wkp, Kf, nullptr, ROWS, 1024, 1024, 0, stream);
        launch_gemm(ap, wvp, Vf, nullptr, ROWS, 1024, 1024, 0, stream);
        // per-(b,h) fragment packs: src offset = b*SEQ*D + h*DH
        launch_pack_a(Qf, Qp, SEQ / 16, 2, 1024, N_HEADS, (long)SEQ * D_MODEL, DH, 65536, BATCH * N_HEADS, stream);
        launch_pack_a(Kf, Kp, SEQ / 16, 2, 1024, N_HEADS, (long)SEQ * D_MODEL, DH, 65536, BATCH * N_HEADS, stream);
        launch_pack_b(Vf, Vp, 4, 32, 1024, N_HEADS, (long)SEQ * D_MODEL, DH, 65536, BATCH * N_HEADS, stream);
        // fused attention -> t0 (reuses Qf region; attention only reads packed frags)
        attn_kernel<<<dim3(SEQ / 16, BATCH * N_HEADS), 256, 16 * SEQ * sizeof(float), stream>>>(
            Qp, Kp, Vp, t0, causal);
        // output projection
        launch_pack_a(t0, ap, ROWS / 16, 32, 1024, 1, 0, 0, 0, 1, stream);
        launch_gemm(ap, wop, outProj, nullptr, ROWS, 1024, 1024, 0, stream);
    };

    // ---- layer 1: masked self-attention ----
    run_attn_layer(X, Wq1p, Wk1p, Wv1p, Wo1p, act1, 1);
    // norm_masked = X + LN(masked)
    ln_residual_kernel<<<ROWS, 256, 0, stream>>>(act1, X, ln_g, ln_b, act2);

    // ---- layer 2: unmasked attention on norm_masked ----
    run_attn_layer(act2, Wq2p, Wk2p, Wv2p, Wo2p, act1, 0);
    // norm_attn2 = attn2 + LN(attn2)
    ln_residual_kernel<<<ROWS, 256, 0, stream>>>(act1, act1, ln_g, ln_b, act2);

    // ---- FFN ----
    launch_pack_a(act2, ap, ROWS / 16, 32, 1024, 1, 0, 0, 0, 1, stream);
    launch_gemm(ap, W1p, t0, b1, ROWS, D_FF, 1024, 1, stream);            // relu(x@W1+b1), 64MB
    launch_pack_a(t0, ap, ROWS / 16, 128, 4096, 1, 0, 0, 0, 1, stream);   // 32MB packed
    launch_gemm(ap, W2p, act1, b2, ROWS, 1024, D_FF, 0, stream);          // ff

    // out = norm_attn2 + LN(ff)
    ln_residual_kernel<<<ROWS, 256, 0, stream>>>(act1, act2, ln_g, ln_b, (float*)d_out);
}
